// ThreeStageFFN_20993800143454
// MI455X (gfx1250) — matrix operand order repaired and checked by reading, not compiled, not run on hardware
//
#include <hip/hip_runtime.h>
#include <hip/hip_bf16.h>
#include <stdint.h>

// Problem constants (from reference setup_inputs)
#define B_    4
#define S_    2048
#define DM_   1024
#define NI_   8192
#define NP_   4096
#define DPV_  512
#define NO_   8192
#define KI_   1024   // max(NI/8, 64)
#define KP_   512    // max(NP/8, 32)
#define KO_   1024   // max(NO/8, 64)

typedef __attribute__((ext_vector_type(16))) __bf16 v16bf;
typedef __attribute__((ext_vector_type(8)))  float  v8f;

union Frag16 { v16bf v; uint4 q[2]; };

// Branch-free exact-erf gelu: Abramowitz-Stegun 7.1.26 (|err| < 1.5e-7),
// one v_rcp_f32 + one v_exp_f32 (hardware TRANS ops) + straight-line FMAs.
// No exec-mask branching, so it pipelines with the WMMA stream.
__device__ __forceinline__ float gelu_exact(float x) {
    float a = fabsf(x) * 0.70710678118654752440f;       // |x|/sqrt(2)
    float t = __builtin_amdgcn_rcpf(fmaf(0.3275911f, a, 1.0f));
    float p = fmaf(t, 1.061405429f, -1.453152027f);
    p = fmaf(t, p, 1.421413741f);
    p = fmaf(t, p, -0.284496736f);
    p = fmaf(t, p, 0.254829592f);
    p = p * t;
    float e = __expf(-a * a);                           // v_exp_f32
    float erf_abs = fmaf(-p, e, 1.0f);
    float erfv = copysignf(erf_abs, x);
    return 0.5f * x * (1.0f + erfv);
}

__device__ __forceinline__ unsigned f2bf_bits(float f) {
    unsigned u = __float_as_uint(f);
    return (u + 0x7fffu + ((u >> 16) & 1u)) >> 16;   // round-to-nearest-even
}

// ---------------------------------------------------------------------------
// f32 -> bf16 conversion (packs 2 per dword)
// ---------------------------------------------------------------------------
__global__ void cvt_bf16_kernel(const float2* __restrict__ src,
                                unsigned* __restrict__ dst, int n2) {
    int i = blockIdx.x * blockDim.x + threadIdx.x;
    if (i < n2) {
        float2 f = src[i];
        dst[i] = f2bf_bits(f.x) | (f2bf_bits(f.y) << 16);
    }
}

// ---------------------------------------------------------------------------
// Stage 1: acts1[b][n] = mean_s gelu( x[b,s,:] . W1[n,:] )   (bf16 WMMA GEMM)
// Block: 256 thr (8 waves). Block tile 128(M=S rows) x 256(N=neurons),
// loops over all S; K staged through LDS in chunks of 32.
// Each wave: 64x64 output (4x4 tiles of v_wmma_f32_16x16x32_bf16), so the
// fragment-load : wmma ratio is 1:2 (B fragments cached in registers and
// reused across 4 A fragments).
// ---------------------------------------------------------------------------
__global__ __launch_bounds__(256) void stage1_wmma_kernel(
    const unsigned short* __restrict__ xb,    // [B][S][DM] bf16 bits
    const unsigned short* __restrict__ w1b,   // [NI][DM]  bf16 bits
    float* __restrict__ acts1)                // [B][NI]
{
    __shared__ alignas(16) unsigned short Alds[128 * 40]; // 128 rows x 32 K (+8 pad)
    __shared__ alignas(16) unsigned short Blds[256 * 40]; // 256 rows x 32 K (+8 pad)
    __shared__ float colsum[256];

    const int tid  = threadIdx.x;
    const int b    = blockIdx.y;
    const int n0   = blockIdx.x * 256;
    const int lane = tid & 31;
    const int wid  = tid >> 5;
    const int wm   = wid & 1;      // 0..1 : 64-row half of M
    const int wn   = wid >> 1;     // 0..3 : 64-col strip of N
    const int l    = lane & 15;
    const int h    = lane >> 4;

    colsum[tid] = 0.0f;
    __syncthreads();

    float cs[4] = {0.f, 0.f, 0.f, 0.f};

    const int fr = tid >> 2;       // fill: row 0..63
    const int fc = tid & 3;        // fill: 16B chunk 0..3

    for (int sb = 0; sb < S_ / 128; ++sb) {
        const int s0 = sb * 128;
        v8f zero = {};
        v8f acc[4][4];
        #pragma unroll
        for (int mt = 0; mt < 4; ++mt)
            #pragma unroll
            for (int nt = 0; nt < 4; ++nt)
                acc[mt][nt] = zero;

        for (int kc = 0; kc < DM_ / 32; ++kc) {
            __syncthreads();   // protect LDS from previous iteration's readers
            // A tile: 128 rows x 64B
            #pragma unroll
            for (int i = 0; i < 2; ++i) {
                int r = fr + i * 64;
                const uint4* pa = (const uint4*)(xb +
                    ((size_t)(b * S_ + s0 + r)) * DM_ + kc * 32 + fc * 8);
                *(uint4*)(&Alds[r * 40 + fc * 8]) = *pa;
            }
            // B tile: 256 rows x 64B
            #pragma unroll
            for (int i = 0; i < 4; ++i) {
                int r = fr + i * 64;
                const uint4* pb = (const uint4*)(w1b +
                    ((size_t)(n0 + r)) * DM_ + kc * 32 + fc * 8);
                *(uint4*)(&Blds[r * 40 + fc * 8]) = *pb;
            }
            __syncthreads();

            // B fragments (lane=N col; lanes<16: K0..15, lanes>=16: K16..31)
            Frag16 bfr[4];
            #pragma unroll
            for (int nt = 0; nt < 4; ++nt) {
                const unsigned short* br = &Blds[(wn * 64 + nt * 16 + l) * 40];
                bfr[nt].q[0] = *(const uint4*)(br + h * 16);
                bfr[nt].q[1] = *(const uint4*)(br + h * 16 + 8);
            }
            #pragma unroll
            for (int mt = 0; mt < 4; ++mt) {
                // A fragment (lane=M row; lanes<16: K0..15, lanes>=16: K16..31)
                Frag16 af;
                const unsigned short* ar = &Alds[(wm * 64 + mt * 16 + l) * 40];
                af.q[0] = *(const uint4*)(ar + h * 16);
                af.q[1] = *(const uint4*)(ar + h * 16 + 8);
                #pragma unroll
                for (int nt = 0; nt < 4; ++nt)
                    acc[mt][nt] = __builtin_amdgcn_wmma_f32_16x16x32_bf16(
                        false, af.v, false, bfr[nt].v, (short)0,
                        acc[mt][nt], false, false);
            }
        }

        // gelu epilogue + column (over-M) partial sums (branch-free).
        // C/D layout: lane n, VGPR r -> (M=r, N=n); lane n+16 -> (M=r+8, N=n)
        #pragma unroll
        for (int mt = 0; mt < 4; ++mt)
            #pragma unroll
            for (int nt = 0; nt < 4; ++nt)
                #pragma unroll
                for (int e = 0; e < 8; ++e)
                    cs[nt] += gelu_exact(acc[mt][nt][e]);
    }

    // combine the two row-halves held in lanes n and n+16
    #pragma unroll
    for (int nt = 0; nt < 4; ++nt) {
        cs[nt] += __shfl_xor(cs[nt], 16);
        if (lane < 16)
            atomicAdd(&colsum[wn * 64 + nt * 16 + l], cs[nt]);  // wm=0/1 merge
    }
    __syncthreads();
    acts1[(size_t)b * NI_ + n0 + tid] = colsum[tid] * (1.0f / (float)S_);
}

// ---------------------------------------------------------------------------
// Exact k-th largest via 32-pass MSB radix select on order-preserving keys.
// One block per batch row. thr[b] = value of the k-th largest element.
// ---------------------------------------------------------------------------
__global__ __launch_bounds__(256) void topk_thresh_kernel(
    const float* __restrict__ vals, int n, int k, float* __restrict__ thr)
{
    __shared__ int red[256];
    const float* v = vals + (size_t)blockIdx.x * n;
    unsigned prefix = 0;
    int above = 0;
    for (int bit = 31; bit >= 0; --bit) {
        unsigned cand_hi = (prefix >> bit) | 1u;
        int cnt = 0;
        for (int i = threadIdx.x; i < n; i += 256) {
            unsigned u = __float_as_uint(v[i]);
            unsigned key = (u & 0x80000000u) ? ~u : (u | 0x80000000u);
            cnt += ((key >> bit) == cand_hi) ? 1 : 0;
        }
        red[threadIdx.x] = cnt;
        __syncthreads();
        for (int s = 128; s > 0; s >>= 1) {
            if (threadIdx.x < s) red[threadIdx.x] += red[threadIdx.x + s];
            __syncthreads();
        }
        int c = red[0];
        __syncthreads();
        if (above + c >= k) prefix |= (1u << bit);
        else above += c;
    }
    if (threadIdx.x == 0) {
        unsigned u = (prefix & 0x80000000u) ? (prefix ^ 0x80000000u) : ~prefix;
        thr[blockIdx.x] = __uint_as_float(u);
    }
}

// input_repr[b][n] = acts[b][n] if in top-k (>= threshold) else 0
__global__ void select_scatter_kernel(const float* __restrict__ acts,
                                      const float* __restrict__ thr,
                                      float* __restrict__ out, int n)
{
    int b = blockIdx.y;
    float t = thr[b];
    for (int i = blockIdx.x * blockDim.x + threadIdx.x; i < n;
         i += gridDim.x * blockDim.x) {
        float a = acts[(size_t)b * n + i];
        out[(size_t)b * n + i] = (a >= t) ? a : 0.0f;
    }
}

// ---------------------------------------------------------------------------
// out[b][r] = gelu( xin[b,:] . W[r,:] ) for all 4 batches at once.
// One wave per row; memory-bound on W (streams once from HBM).
// ---------------------------------------------------------------------------
__global__ __launch_bounds__(256) void gemv4_gelu_kernel(
    const float* __restrict__ W, const float* __restrict__ xin,
    float* __restrict__ out, int rows, int ncol)
{
    const int lane = threadIdx.x & 31;
    const int r = blockIdx.x * 8 + (threadIdx.x >> 5);
    if (r >= rows) return;
    const float4* wr = (const float4*)(W + (size_t)r * ncol);
    const float4* x0 = (const float4*)xin;
    const int nq = ncol >> 2;
    float a0 = 0.f, a1 = 0.f, a2 = 0.f, a3 = 0.f;
    for (int i = lane; i < nq; i += 32) {
        __builtin_prefetch(wr + i + 64);   // global_prefetch_b8
        float4 w = wr[i];
        float4 v0 = x0[i], v1 = x0[nq + i], v2 = x0[2 * nq + i], v3 = x0[3 * nq + i];
        a0 = fmaf(w.x, v0.x, fmaf(w.y, v0.y, fmaf(w.z, v0.z, fmaf(w.w, v0.w, a0))));
        a1 = fmaf(w.x, v1.x, fmaf(w.y, v1.y, fmaf(w.z, v1.z, fmaf(w.w, v1.w, a1))));
        a2 = fmaf(w.x, v2.x, fmaf(w.y, v2.y, fmaf(w.z, v2.z, fmaf(w.w, v2.w, a2))));
        a3 = fmaf(w.x, v3.x, fmaf(w.y, v3.y, fmaf(w.z, v3.z, fmaf(w.w, v3.w, a3))));
    }
    #pragma unroll
    for (int off = 16; off > 0; off >>= 1) {
        a0 += __shfl_down(a0, off);
        a1 += __shfl_down(a1, off);
        a2 += __shfl_down(a2, off);
        a3 += __shfl_down(a3, off);
    }
    if (lane == 0) {
        out[r]            = gelu_exact(a0);
        out[rows + r]     = gelu_exact(a1);
        out[2 * rows + r] = gelu_exact(a2);
        out[3 * rows + r] = gelu_exact(a3);
    }
}

// ---------------------------------------------------------------------------
// Stage 2 tail: softmax over selected process neurons, aggregate values.
// agg[b][d] = sum_p softmax_w(p) * PV[p][d], selected = acts >= thr.
// ---------------------------------------------------------------------------
__global__ __launch_bounds__(256) void softmax_agg_kernel(
    const float* __restrict__ pacts, const float* __restrict__ thr,
    const float* __restrict__ PV, float* __restrict__ agg)
{
    __shared__ float sw[NP_];
    __shared__ float red[256];
    const int tid = threadIdx.x;
    const int b = blockIdx.x;
    const float t = thr[b];

    float mx = -3.4e38f;   // global max == max of top-k
    for (int i = tid; i < NP_; i += 256)
        mx = fmaxf(mx, pacts[(size_t)b * NP_ + i]);
    red[tid] = mx; __syncthreads();
    for (int s = 128; s > 0; s >>= 1) {
        if (tid < s) red[tid] = fmaxf(red[tid], red[tid + s]);
        __syncthreads();
    }
    mx = red[0]; __syncthreads();

    float se = 0.f;
    for (int i = tid; i < NP_; i += 256) {
        float a = pacts[(size_t)b * NP_ + i];
        float w = (a >= t) ? expf(a - mx) : 0.0f;
        sw[i] = w;
        se += w;
    }
    red[tid] = se; __syncthreads();
    for (int s = 128; s > 0; s >>= 1) {
        if (tid < s) red[tid] += red[tid + s];
        __syncthreads();
    }
    se = red[0]; __syncthreads();
    const float inv = 1.0f / se;

    float acc0 = 0.f, acc1 = 0.f;      // DPV_=512, 2 dims/thread
    for (int p = 0; p < NP_; ++p) {
        float w = sw[p];
        if (w != 0.0f) {
            const float* row = PV + (size_t)p * DPV_;
            acc0 = fmaf(w, row[tid], acc0);
            acc1 = fmaf(w, row[tid + 256], acc1);
        }
    }
    agg[(size_t)b * DPV_ + tid]       = acc0 * inv;
    agg[(size_t)b * DPV_ + tid + 256] = acc1 * inv;
}

// ---------------------------------------------------------------------------
// Stage 3 tail: oseq[b][m] = sum over selected o of act_o * OP[o][m]
// ---------------------------------------------------------------------------
__global__ __launch_bounds__(256) void out_agg_kernel(
    const float* __restrict__ oacts, const float* __restrict__ thr,
    const float* __restrict__ OP, float* __restrict__ oseq)
{
    __shared__ float sw[NO_];          // 32 KB of the 320 KB LDS
    const int tid = threadIdx.x;
    const int b = blockIdx.x;
    const float t = thr[b];
    for (int i = tid; i < NO_; i += 256) {
        float a = oacts[(size_t)b * NO_ + i];
        sw[i] = (a >= t) ? a : 0.0f;
    }
    __syncthreads();
    float acc[4] = {0.f, 0.f, 0.f, 0.f};   // DM_=1024, 4 dims/thread
    for (int o = 0; o < NO_; ++o) {
        float w = sw[o];
        if (w != 0.0f) {
            const float* row = OP + (size_t)o * DM_;
            #pragma unroll
            for (int j = 0; j < 4; ++j)
                acc[j] = fmaf(w, row[tid + j * 256], acc[j]);
        }
    }
    #pragma unroll
    for (int j = 0; j < 4; ++j)
        oseq[(size_t)b * DM_ + tid + j * 256] = acc[j];
}

// output[b][s][m] = oseq[b][m]   (vectorized broadcast store)
__global__ void bcast_kernel(const float4* __restrict__ oseq, float4* __restrict__ out)
{
    const int dm4 = DM_ / 4;
    size_t idx = (size_t)blockIdx.x * blockDim.x + threadIdx.x;
    int m4 = (int)(idx % dm4);
    int b  = (int)(idx / ((size_t)S_ * dm4));
    out[idx] = oseq[b * dm4 + m4];
}

// ---------------------------------------------------------------------------
extern "C" void kernel_launch(void* const* d_in, const int* in_sizes, int n_in,
                              void* d_out, int out_size, void* d_ws, size_t ws_size,
                              hipStream_t stream) {
    (void)in_sizes; (void)n_in; (void)out_size; (void)ws_size;
    const float* x   = (const float*)d_in[0];   // [B,S,DM]
    const float* ip  = (const float*)d_in[1];   // [NI,DM]
    const float* piw = (const float*)d_in[2];   // [NP,NI]
    const float* pv  = (const float*)d_in[3];   // [NP,DPV]
    const float* oiw = (const float*)d_in[4];   // [NO,DPV]
    const float* op  = (const float*)d_in[5];   // [NO,DM]
    float* out = (float*)d_out;

    char* ws = (char*)d_ws;
    size_t off = 0;
    auto alloc = [&](size_t bytes) -> void* {
        void* p = ws + off;
        off += (bytes + 255) & ~(size_t)255;
        return p;
    };
    unsigned short* xb    = (unsigned short*)alloc((size_t)B_ * S_ * DM_ * 2); // 16 MB
    unsigned short* w1b   = (unsigned short*)alloc((size_t)NI_ * DM_ * 2);     // 16 MB
    float*          acts1 = (float*)alloc((size_t)B_ * NI_ * 4);
    float*          repr  = (float*)alloc((size_t)B_ * NI_ * 4);
    float*          pacts = (float*)alloc((size_t)B_ * NP_ * 4);
    float*          agg   = (float*)alloc((size_t)B_ * DPV_ * 4);
    float*          oacts = (float*)alloc((size_t)B_ * NO_ * 4);
    float*          oseq  = (float*)alloc((size_t)B_ * DM_ * 4);
    float*          thr   = (float*)alloc(64);   // thr[0..3], [4..7], [8..11]

    // bf16 conversion of x and input_patterns
    int n2x = B_ * S_ * DM_ / 2;
    cvt_bf16_kernel<<<(n2x + 255) / 256, 256, 0, stream>>>(
        (const float2*)x, (unsigned*)xb, n2x);
    int n2w = NI_ * DM_ / 2;
    cvt_bf16_kernel<<<(n2w + 255) / 256, 256, 0, stream>>>(
        (const float2*)ip, (unsigned*)w1b, n2w);

    // Stage 1: fused WMMA GEMM + gelu + mean-over-S
    stage1_wmma_kernel<<<dim3(NI_ / 256, B_), 256, 0, stream>>>(xb, w1b, acts1);
    topk_thresh_kernel<<<B_, 256, 0, stream>>>(acts1, NI_, KI_, thr);
    select_scatter_kernel<<<dim3(32, B_), 256, 0, stream>>>(acts1, thr, repr, NI_);

    // Stage 2: process neurons (memory-bound GEMV) + softmax aggregation
    gemv4_gelu_kernel<<<NP_ / 8, 256, 0, stream>>>(piw, repr, pacts, NP_, NI_);
    topk_thresh_kernel<<<B_, 256, 0, stream>>>(pacts, NP_, KP_, thr + 4);
    softmax_agg_kernel<<<B_, 256, 0, stream>>>(pacts, thr + 4, pv, agg);

    // Stage 3: output neurons + gather-weighted pattern sum + broadcast
    gemv4_gelu_kernel<<<NO_ / 8, 256, 0, stream>>>(oiw, agg, oacts, NO_, DPV_);
    topk_thresh_kernel<<<B_, 256, 0, stream>>>(oacts, NO_, KO_, thr + 8);
    out_agg_kernel<<<B_, 256, 0, stream>>>(oacts, thr + 8, op, oseq);
    bcast_kernel<<<(B_ * S_ * DM_ / 4) / 256, 256, 0, stream>>>(
        (const float4*)oseq, (float4*)out);
}
